// MoE_83511344103771
// MI455X (gfx1250) — matrix operand order repaired and checked by reading, not compiled, not run on hardware
//
#include <hip/hip_runtime.h>
#include <math.h>

// ---------------------------------------------------------------------------
// MoE-GCN forward for MI455X (gfx1250).
// Dense GEMMs (94 GFLOP fp32) done as split-bf16 (hi/lo) WMMA with f32 accum:
//   A*B ~= Ahi*Bhi + Alo*Bhi + Ahi*Blo   (3x v_wmma_f32_16x16x32_bf16)
// Graph scatter/gather stays in fp32 through L2 (working set < 192MB L2).
// ---------------------------------------------------------------------------

#define G_GRAPHS 256
#define NEXP     8
#define DDIM     256
#define KTOP     4

typedef __attribute__((ext_vector_type(16))) __bf16        v16bf;
typedef __attribute__((ext_vector_type(8)))  float         v8f;
typedef __attribute__((ext_vector_type(8)))  unsigned int  v8u;

__device__ __forceinline__ unsigned short f32_to_bf16(float f) {
  unsigned int u = __float_as_uint(f);
  u += 0x7fffu + ((u >> 16) & 1u);      // round-to-nearest-even
  return (unsigned short)(u >> 16);
}
__device__ __forceinline__ float bf16_to_f32(unsigned short h) {
  return __uint_as_float(((unsigned int)h) << 16);
}
__device__ __forceinline__ void atomic_add_f32(float* p, float v) {
  __hip_atomic_fetch_add(p, v, __ATOMIC_RELAXED, __HIP_MEMORY_SCOPE_AGENT);
}

// Packed A-fragment index (WMMA 16x16x32 bf16 A layout, wave32):
// lane<16: row=lane,   K = {0..7, 16..23} of the 32-block
// lane>=16: row=lane-16,K = {8..15,24..31}
__device__ __forceinline__ int packA_idx(int row, int d) {
  int mtile = row >> 4, ml = row & 15, kb = d >> 5, kl = d & 31;
  int lane = ml + (((kl >> 3) & 1) << 4);
  int t    = (kl & 7) + ((kl >> 4) << 3);
  return ((mtile * 8 + kb) * 32 + lane) * 16 + t;
}

// ------------------------------ utility kernels ----------------------------

__global__ void fill_f32(float* p, float v, int n) {
  int i = blockIdx.x * blockDim.x + threadIdx.x;
  if (i < n) p[i] = v;
}

__global__ void deg_kernel(const int* __restrict__ ei, float* __restrict__ deg, int E) {
  int e = blockIdx.x * blockDim.x + threadIdx.x;
  if (e < E) atomic_add_f32(&deg[ei[E + e]], 1.0f);
}

__global__ void rsqrt_kernel(float* d, int n) {
  int i = blockIdx.x * blockDim.x + threadIdx.x;
  if (i < n) d[i] = rsqrtf(d[i]);          // deg >= 1 (self loop)
}

// segment-sum pool: out[(batch[row]*ne + e)*256 + d] += src[row*256+d]
__global__ void pool_kernel(const float* __restrict__ src, const int* __restrict__ batch,
                            float* __restrict__ out, int ne, int e, int total) {
  int i = blockIdx.x * blockDim.x + threadIdx.x;
  if (i >= total) return;
  int row = i >> 8, d = i & 255;
  atomic_add_f32(&out[((size_t)batch[row] * ne + e) * DDIM + d], src[i]);
}

// ------------------------- operand packing kernels -------------------------

// expert_W [36][256][256] (k-major rows, n contiguous) -> packed B fragments:
// per 32x16 (KxN) tile: lane<16 holds col n, K {0..7,16..23} ; lane>=16 K {8..15,24..31}
__global__ void pack_w_kernel(const float* __restrict__ W,
                              unsigned short* __restrict__ whi,
                              unsigned short* __restrict__ wlo, int total) {
  int i = blockIdx.x * blockDim.x + threadIdx.x;
  if (i >= total) return;
  int l = i >> 16, k = (i >> 8) & 255, n = i & 255;
  float v = W[i];
  unsigned short h = f32_to_bf16(v);
  unsigned short lo = f32_to_bf16(v - bf16_to_f32(h));
  int kb = k >> 5, kl = k & 31, nb = n >> 4, nl = n & 15;
  int lane = nl + (((kl >> 3) & 1) << 4);
  int t = (kl & 7) + ((kl >> 4) << 3);
  size_t idx = ((((size_t)l * 8 + kb) * 16 + nb) * 32 + lane) * 16 + t;
  whi[idx] = h; wlo[idx] = lo;
}

__global__ void pack_x_kernel(const float* __restrict__ x,
                              unsigned short* __restrict__ ahi,
                              unsigned short* __restrict__ alo, int total) {
  int i = blockIdx.x * blockDim.x + threadIdx.x;
  if (i >= total) return;
  float v = x[i];
  unsigned short h = f32_to_bf16(v);
  unsigned short lo = f32_to_bf16(v - bf16_to_f32(h));
  int idx = packA_idx(i >> 8, i & 255);
  ahi[idx] = h; alo[idx] = lo;
}

// ------------------------------- WMMA GEMM ---------------------------------
// C[row, :] = dis[row] * (A @ W) ; A: mtiles*16 x 256 packed split-bf16,
// W: 256x256 packed split-bf16 (one layer). Block = 8 waves = 64x128 tile.
__global__ __launch_bounds__(256) void wmma_gemm_split_bf16(
    const unsigned short* __restrict__ ahi, const unsigned short* __restrict__ alo,
    const unsigned short* __restrict__ bhi, const unsigned short* __restrict__ blo,
    const float* __restrict__ dis, float* __restrict__ out, int mtiles) {
  const int lane = threadIdx.x & 31;
  const int wid  = threadIdx.x >> 5;
  const int rg = wid & 3;                 // 4 row-groups of 16 rows
  const int cg = wid >> 2;                // 2 col-groups of 64 cols
  const int mtile = blockIdx.x * 4 + rg;
  const int mc = mtile < mtiles ? mtile : (mtiles - 1);   // clamp, keep EXEC full
  const int nb0 = blockIdx.y * 8 + cg * 4;                // first 16-col tile

  v8f acc[4] = {};

  for (int kb = 0; kb < 8; ++kb) {
    const int abase = ((mc * 8 + kb) * 32 + lane) * 16;
    v16bf a_hi = __builtin_bit_cast(v16bf, *(const v8u*)(ahi + abase));
    v16bf a_lo = __builtin_bit_cast(v16bf, *(const v8u*)(alo + abase));
#pragma unroll
    for (int t = 0; t < 4; ++t) {
      const int bbase = ((kb * 16 + nb0 + t) * 32 + lane) * 16;
      v16bf b_hi = __builtin_bit_cast(v16bf, *(const v8u*)(bhi + bbase));
      v16bf b_lo = __builtin_bit_cast(v16bf, *(const v8u*)(blo + bbase));
      acc[t] = __builtin_amdgcn_wmma_f32_16x16x32_bf16(false, a_hi, false, b_hi,
                                                       (short)0, acc[t], false, false);
      acc[t] = __builtin_amdgcn_wmma_f32_16x16x32_bf16(false, a_lo, false, b_hi,
                                                       (short)0, acc[t], false, false);
      acc[t] = __builtin_amdgcn_wmma_f32_16x16x32_bf16(false, a_hi, false, b_lo,
                                                       (short)0, acc[t], false, false);
    }
  }

  if (mtile < mtiles) {
    const int colb = lane & 15;
    const int rowoff = (lane >> 4) << 3;  // lanes 16-31 hold rows m+8..m+15
#pragma unroll
    for (int r = 0; r < 8; ++r) {
      const int row = mtile * 16 + rowoff + r;
      const float s = dis[row];
      float* orow = out + (size_t)row * DDIM;
#pragma unroll
      for (int t = 0; t < 4; ++t)
        orow[(nb0 + t) * 16 + colb] = acc[t][r] * s;
    }
  }
}

// ------------------------------ graph scatter ------------------------------
// acc[dst] += tmp[src] for every edge; one wave per edge, float4 per lane.
__global__ __launch_bounds__(256) void scatter_kernel(const int* __restrict__ ei,
                                                      const float* __restrict__ tmp,
                                                      float* __restrict__ acc, int E) {
  int gw = (blockIdx.x * blockDim.x + threadIdx.x) >> 5;
  int lane = threadIdx.x & 31;
  if (gw >= E) return;
  int src = ei[gw], dst = ei[E + gw];
  const float4* srow = (const float4*)(tmp + (size_t)src * DDIM);
  float* drow = acc + (size_t)dst * DDIM;
#pragma unroll
  for (int it = 0; it < 2; ++it) {
    int q = it * 32 + lane;
    float4 v = srow[q];
    atomic_add_f32(drow + q * 4 + 0, v.x);
    atomic_add_f32(drow + q * 4 + 1, v.y);
    atomic_add_f32(drow + q * 4 + 2, v.z);
    atomic_add_f32(drow + q * 4 + 3, v.w);
  }
}

// h' = relu(dis[i]*(acc + tmp_self) + b); also emit packed split-bf16 for next GEMM
__global__ void finalize_kernel(const float* __restrict__ acc, const float* __restrict__ tmp,
                                const float* __restrict__ dis, const float* __restrict__ bias,
                                float* __restrict__ h, unsigned short* __restrict__ ahi,
                                unsigned short* __restrict__ alo, int total) {
  int i = blockIdx.x * blockDim.x + threadIdx.x;
  if (i >= total) return;
  int row = i >> 8, d = i & 255;
  float v = dis[row] * (acc[i] + tmp[i]) + bias[d];
  v = fmaxf(v, 0.0f);
  h[i] = v;
  unsigned short hv = f32_to_bf16(v);
  unsigned short lv = f32_to_bf16(v - bf16_to_f32(hv));
  int idx = packA_idx(row, d);
  ahi[idx] = hv; alo[idx] = lv;
}

// ------------------------------ gating / loss ------------------------------

__device__ __forceinline__ float ncdf(float z) {
  return 0.5f * (1.0f + erff(z * 0.70710678118654752440f));
}

__global__ void gate_kernel(const float* __restrict__ xp, const float* __restrict__ wg,
                            const float* __restrict__ wn, const float* __restrict__ noise,
                            float* __restrict__ gates, float* __restrict__ loadc) {
  int g = threadIdx.x;
  if (g >= G_GRAPHS) return;
  float clean[NEXP] = {}, nz[NEXP] = {};
  for (int d = 0; d < DDIM; ++d) {
    float xv = xp[(size_t)g * DDIM + d];
#pragma unroll
    for (int e = 0; e < NEXP; ++e) {
      clean[e] += xv * wg[d * NEXP + e];
      nz[e]    += xv * wn[d * NEXP + e];
    }
  }
  float stdv[NEXP], noisy[NEXP];
#pragma unroll
  for (int e = 0; e < NEXP; ++e) {
    float s = fmaxf(nz[e], 0.0f) + log1pf(expf(-fabsf(nz[e])));  // softplus
    stdv[e] = s + 0.01f;
    noisy[e] = clean[e] + noise[g * NEXP + e] * stdv[e];
  }
  // top-(K+1) selection, strict > keeps lowest index on ties (jax semantics)
  float tv[KTOP + 1]; int ti[KTOP + 1]; bool used[NEXP] = {};
  for (int r = 0; r < KTOP + 1; ++r) {
    float best = -INFINITY; int bi = 0;
    for (int e = 0; e < NEXP; ++e)
      if (!used[e] && noisy[e] > best) { best = noisy[e]; bi = e; }
    used[bi] = true; tv[r] = best; ti[r] = bi;
  }
  float m = tv[0], s = 0.0f, ex[KTOP];
#pragma unroll
  for (int r = 0; r < KTOP; ++r) { ex[r] = expf(tv[r] - m); s += ex[r]; }
  float gz[NEXP] = {};
#pragma unroll
  for (int r = 0; r < KTOP; ++r) gz[ti[r]] = ex[r] / s;
  float thr_in = tv[KTOP], thr_out = tv[KTOP - 1];
#pragma unroll
  for (int e = 0; e < NEXP; ++e) {
    gates[g * NEXP + e] = gz[e];
    float thr = (noisy[e] > thr_in) ? thr_in : thr_out;
    loadc[g * NEXP + e] = ncdf((clean[e] - thr) / stdv[e]);
  }
}

__device__ __forceinline__ float cv_sq8(const float* v) {
  float m = 0.0f;
  for (int e = 0; e < NEXP; ++e) m += v[e];
  m *= (1.0f / NEXP);
  float var = 0.0f;
  for (int e = 0; e < NEXP; ++e) { float d = v[e] - m; var += d * d; }
  var *= (1.0f / (NEXP - 1));
  return var / (m * m + 1e-10f);
}

__global__ void loss_kernel(const float* __restrict__ gates, const float* __restrict__ loadc,
                            float* __restrict__ out) {
  __shared__ float imp[NEXP], ld[NEXP];
  int e = threadIdx.x;
  if (e < NEXP) {
    float si = 0.0f, sl = 0.0f;
    for (int g = 0; g < G_GRAPHS; ++g) { si += gates[g * NEXP + e]; sl += loadc[g * NEXP + e]; }
    imp[e] = si; ld[e] = sl;
  }
  __syncthreads();
  if (threadIdx.x == 0) out[0] = (cv_sq8(imp) + cv_sq8(ld)) * 0.001f;
}

__global__ void combine_kernel(const float* __restrict__ gates, const float* __restrict__ eout,
                               float* __restrict__ y, int total) {
  int i = blockIdx.x * blockDim.x + threadIdx.x;
  if (i >= total) return;
  int g = i >> 8, d = i & 255;
  float s = 0.0f;
#pragma unroll
  for (int e = 0; e < NEXP; ++e)
    s += gates[g * NEXP + e] * eout[((size_t)g * NEXP + e) * DDIM + d];
  y[i] = s;
}

// --------------------------------- driver ----------------------------------

extern "C" void kernel_launch(void* const* d_in, const int* in_sizes, int n_in,
                              void* d_out, int out_size, void* d_ws, size_t ws_size,
                              hipStream_t stream) {
  const float* x     = (const float*)d_in[0];
  const int*   ei    = (const int*)d_in[1];
  const int*   batch = (const int*)d_in[2];
  const float* noise = (const float*)d_in[3];
  const float* eW    = (const float*)d_in[4];
  const float* eb    = (const float*)d_in[5];
  const float* wg    = (const float*)d_in[6];
  const float* wn    = (const float*)d_in[7];
  float* y = (float*)d_out;

  const int N = in_sizes[0] / DDIM;       // 20000
  const int E = in_sizes[1] / 2;          // 320000
  const int T = N * DDIM;
  const int TOTAL_L = in_sizes[5] / DDIM; // 36
  const int mtiles = N / 16;              // 1250

  size_t off = 0;
  char* base = (char*)d_ws;
  auto carve = [&](size_t bytes) -> char* {
    char* p = base + off;
    off += (bytes + 255) & ~(size_t)255;
    return p;
  };
  float* dis   = (float*)carve((size_t)N * 4);
  float* xp    = (float*)carve((size_t)G_GRAPHS * DDIM * 4);
  float* gates = (float*)carve((size_t)G_GRAPHS * NEXP * 4);
  float* loadc = (float*)carve((size_t)G_GRAPHS * NEXP * 4);
  unsigned short* wpk_hi = (unsigned short*)carve((size_t)TOTAL_L * 65536 * 2);
  unsigned short* wpk_lo = (unsigned short*)carve((size_t)TOTAL_L * 65536 * 2);
  unsigned short* xpk_hi = (unsigned short*)carve((size_t)T * 2);
  unsigned short* xpk_lo = (unsigned short*)carve((size_t)T * 2);
  unsigned short* hpk_hi = (unsigned short*)carve((size_t)T * 2);
  unsigned short* hpk_lo = (unsigned short*)carve((size_t)T * 2);
  float* h   = (float*)carve((size_t)T * 4);
  float* tmp = (float*)carve((size_t)T * 4);
  float* acc = (float*)carve((size_t)T * 4);
  float* eout = (float*)carve((size_t)G_GRAPHS * NEXP * DDIM * 4);

  auto cdiv = [](int a, int b) { return (a + b - 1) / b; };
  const int BS = 256;

  // ---- setup: degrees, pools, operand packing, gating ----
  fill_f32<<<cdiv(N, BS), BS, 0, stream>>>(dis, 1.0f, N);             // self loop
  fill_f32<<<cdiv(G_GRAPHS * DDIM, BS), BS, 0, stream>>>(xp, 0.0f, G_GRAPHS * DDIM);
  fill_f32<<<cdiv(G_GRAPHS * NEXP * DDIM, BS), BS, 0, stream>>>(eout, 0.0f, G_GRAPHS * NEXP * DDIM);
  deg_kernel<<<cdiv(E, BS), BS, 0, stream>>>(ei, dis, E);
  rsqrt_kernel<<<cdiv(N, BS), BS, 0, stream>>>(dis, N);
  pool_kernel<<<cdiv(T, BS), BS, 0, stream>>>(x, batch, xp, 1, 0, T);
  pack_w_kernel<<<cdiv(TOTAL_L * 65536, BS), BS, 0, stream>>>(eW, wpk_hi, wpk_lo, TOTAL_L * 65536);
  pack_x_kernel<<<cdiv(T, BS), BS, 0, stream>>>(x, xpk_hi, xpk_lo, T);
  gate_kernel<<<1, 256, 0, stream>>>(xp, wg, wn, noise, gates, loadc);
  loss_kernel<<<1, 32, 0, stream>>>(gates, loadc, y + (size_t)G_GRAPHS * DDIM);

  // ---- expert chains: nl = 1..8 GCN layers each ----
  dim3 ggrid(cdiv(mtiles, 4), 2);
  int lofs = 0;
  for (int e = 0; e < NEXP; ++e) {
    int nl = e + 1;
    const unsigned short* cahi = xpk_hi;
    const unsigned short* calo = xpk_lo;
    for (int j = 0; j < nl; ++j) {
      int l = lofs + j;
      wmma_gemm_split_bf16<<<ggrid, 256, 0, stream>>>(
          cahi, calo, wpk_hi + (size_t)l * 65536, wpk_lo + (size_t)l * 65536,
          dis, tmp, mtiles);
      fill_f32<<<cdiv(T, BS), BS, 0, stream>>>(acc, 0.0f, T);
      scatter_kernel<<<cdiv(E * 32, BS), BS, 0, stream>>>(ei, tmp, acc, E);
      finalize_kernel<<<cdiv(T, BS), BS, 0, stream>>>(acc, tmp, dis, eb + (size_t)l * DDIM,
                                                      h, hpk_hi, hpk_lo, T);
      cahi = hpk_hi; calo = hpk_lo;
    }
    lofs += nl;
    pool_kernel<<<cdiv(T, BS), BS, 0, stream>>>(h, batch, eout, NEXP, e, T);
  }

  // ---- combine: y = sum_e gates * expert_out ----
  combine_kernel<<<cdiv(G_GRAPHS * DDIM, BS), BS, 0, stream>>>(gates, eout, y, G_GRAPHS * DDIM);
}
